// IngredientLabeller_41635412967830
// MI455X (gfx1250) — matrix operand order repaired and checked by reading, not compiled, not run on hardware
//
#include <hip/hip_runtime.h>

#define DEV __device__ __forceinline__

typedef __attribute__((ext_vector_type(16))) __bf16 v16bf;
typedef __attribute__((ext_vector_type(8)))  float  v8f;

struct U4 { unsigned x, y, z, w; };

union FragBF {
  v16bf v;
  unsigned short u[16];
  U4 q[2];
};

DEV unsigned short f2bf(float f) {
  unsigned u = __float_as_uint(f);
  u += 0x7FFFu + ((u >> 16) & 1u);       // round-to-nearest-even
  return (unsigned short)(u >> 16);
}
DEV float bf2f(unsigned short h) { return __uint_as_float(((unsigned)h) << 16); }
DEV float sigf(float x) { return 1.0f / (1.0f + __expf(-x)); }
DEV float wred(float x) {
  #pragma unroll
  for (int o = 16; o > 0; o >>= 1) x += __shfl_down(x, o, 32);
  return x;
}

// Async global -> LDS copy (gfx1250, ASYNCcnt-tracked, bypasses VGPRs).
// LDS byte offset = low 32 bits of the generic pointer (aperture layout).
DEV void async_copy_b128(void* lds_ptr, const void* gptr) {
  unsigned lds_off = (unsigned)(uintptr_t)lds_ptr;
  unsigned long long ga = (unsigned long long)(uintptr_t)gptr;
  asm volatile("global_load_async_to_lds_b128 %0, %1, off"
               :
               : "v"(lds_off), "v"(ga)
               : "memory");
}
DEV void wait_asynccnt0() {
#if __has_builtin(__builtin_amdgcn_s_wait_asynccnt)
  __builtin_amdgcn_s_wait_asynccnt(0);
#else
  asm volatile("s_wait_asynccnt 0x0" : : : "memory");
#endif
}

// ---------------- small prep kernels ----------------

__global__ void k_zero(float* acc) { if (threadIdx.x < 2) acc[threadIdx.x] = 0.f; }

// featmean[b*1024+c] = mean over 768 features; one wave per (b,c)
__global__ void k_featmean(const float* __restrict__ ce, float* __restrict__ fm) {
  int lane = threadIdx.x & 31, w = threadIdx.x >> 5;
  int bc = blockIdx.x * 8 + w;                     // 0..32767
  const float* p = ce + (size_t)bc * 768;
  float s = 0.f;
  for (int k = lane; k < 768; k += 32) s += p[k];
  s = wred(s);
  if (lane == 0) fm[bc] = s * (1.0f / 768.0f);
}

// val[b*256+s] = mean of featmean over char span (0 when padded)
__global__ void k_val(const int* __restrict__ spans, const float* __restrict__ fm,
                      float* __restrict__ val) {
  int r = blockIdx.x * 256 + threadIdx.x;
  int st = spans[2 * r], en = spans[2 * r + 1];
  float v = 0.f;
  if (st >= 0) {
    int e = en < 0 ? 0 : en;
    int n = e - st + 1; if (n < 1) n = 1;
    int b = r >> 8;
    float t = 0.f;
    for (int c = st; c <= e; ++c) t += fm[b * 1024 + c];
    v = t / (float)n;
  }
  val[r] = v;
}

__global__ void k_cvt_x(const float* __restrict__ x, unsigned short* __restrict__ xb, int n) {
  int i = blockIdx.x * 256 + threadIdx.x;
  if (i < n) xb[i] = f2bf(x[i]);
}

// bf16-convert Wih[:, :768]; wsum[j] = sum Wih[j,768:]; biasc[j] = bih[j]+bhh[j]
__global__ void k_prep_wih(const float* __restrict__ Wf, const float* __restrict__ Wb,
                           const float* __restrict__ bihf, const float* __restrict__ bhhf,
                           const float* __restrict__ bihb, const float* __restrict__ bhhb,
                           unsigned short* __restrict__ WihBf,
                           float* __restrict__ wsum, float* __restrict__ biasc) {
  int lane = threadIdx.x & 31, w = threadIdx.x >> 5;
  int g = blockIdx.x * 8 + w;                      // 0..2047
  int dir = g >> 10, j = g & 1023;
  const float* W  = dir ? Wb : Wf;
  const float* bi = dir ? bihb : bihf;
  const float* bh = dir ? bhhb : bhhf;
  unsigned short* Wo = WihBf + (size_t)dir * 1024 * 768;
  const float* row = W + (size_t)j * 2816;
  for (int k = lane; k < 768; k += 32) Wo[j * 768 + k] = f2bf(row[k]);
  float s = 0.f;
  for (int k = 768 + lane; k < 2816; k += 32) s += row[k];
  s = wred(s);
  if (lane == 0) { wsum[dir * 1024 + j] = s; biasc[dir * 1024 + j] = bi[j] + bh[j]; }
}

__global__ void k_cvt_whh(const float* __restrict__ Wf, const float* __restrict__ Wb,
                          unsigned short* __restrict__ out) {
  int i = blockIdx.x * 256 + threadIdx.x;          // 0..524287
  int dir = i >> 18;
  int rem = i & 262143;
  out[i] = f2bf((dir ? Wb : Wf)[rem]);
}

// ---------------- pre-gate GEMM: [8192 x 768] x [768 x 1024] per direction ----------------
// block tile 128x64, 8 waves (4x2), 32x32 per wave = 2x2 WMMA tiles, K step 32.
// Double-buffered LDS tiles filled by async global->LDS (ASYNCcnt), overlapped with WMMA.

__global__ __launch_bounds__(256)
void k_pregemm(const unsigned short* __restrict__ Xbf,
               const unsigned short* __restrict__ WihBf,
               const float* __restrict__ wsum, const float* __restrict__ biasc,
               const float* __restrict__ val, float* __restrict__ preG) {
  __shared__ unsigned short As[2][128][40];        // padded rows (bank spread)
  __shared__ unsigned short Bs[2][64][40];         // Bs[n][k], n = gate col, k = reduction
  int tid = threadIdx.x, lane = tid & 31, w = tid >> 5;
  int dir = blockIdx.z;
  int rowBase = blockIdx.y * 128;
  int colBase = blockIdx.x * 64;
  const unsigned short* Wd = WihBf + (size_t)dir * (1024 * 768);
  int warpM = w & 3, warpN = w >> 2;

  v8f zero = {0, 0, 0, 0, 0, 0, 0, 0};
  v8f acc[2][2];
  acc[0][0] = zero; acc[0][1] = zero; acc[1][0] = zero; acc[1][1] = zero;

  auto stage = [&](int buf, int kk) {
    // A: 128 rows x 4 x 16B = 512 async b128 ops (2 per thread)
    for (int i = tid; i < 512; i += 256) {
      int r = i >> 2, s = i & 3;
      async_copy_b128(&As[buf][r][s * 8],
                      &Xbf[(size_t)(rowBase + r) * 768 + kk + s * 8]);
    }
    // B: 64 rows x 4 x 16B = 256 async b128 ops (1 per thread)
    {
      int n = tid >> 2, s = tid & 3;
      async_copy_b128(&Bs[buf][n][s * 8],
                      &Wd[(size_t)(colBase + n) * 768 + kk + s * 8]);
    }
  };

  stage(0, 0);                                     // prime the pipeline

  for (int kt = 0; kt < 24; ++kt) {                // 768 / 32
    int buf = kt & 1;
    wait_asynccnt0();                              // this wave's fills done
    __syncthreads();                               // all waves' fills visible; prev reads retired
    if (kt + 1 < 24) stage(buf ^ 1, (kt + 1) * 32);

    #pragma unroll
    for (int mt = 0; mt < 2; ++mt) {
      FragBF a;
      const U4* pa = (const U4*)&As[buf][warpM * 32 + mt * 16 + (lane & 15)][0];
      a.q[0] = pa[lane >> 4];                      // K 0-7 / 8-15
      a.q[1] = pa[2 + (lane >> 4)];                // K 16-23 / 24-31
      #pragma unroll
      for (int nt = 0; nt < 2; ++nt) {
        FragBF b;
        const U4* pb = (const U4*)&Bs[buf][warpN * 32 + nt * 16 + (lane & 15)][0];
        b.q[0] = pb[lane >> 4];                    // K 0-7 / 8-15
        b.q[1] = pb[2 + (lane >> 4)];              // K 16-23 / 24-31
        acc[mt][nt] = __builtin_amdgcn_wmma_f32_16x16x32_bf16(
            false, a.v, false, b.v, (short)0, acc[mt][nt], false, false);
      }
    }
  }

  const float* ws = wsum + dir * 1024;
  const float* bc = biasc + dir * 1024;
  float* out = preG + (size_t)dir * (8192ull * 1024ull);
  #pragma unroll
  for (int mt = 0; mt < 2; ++mt)
    #pragma unroll
    for (int nt = 0; nt < 2; ++nt)
      #pragma unroll
      for (int vr = 0; vr < 8; ++vr) {
        int r = rowBase + warpM * 32 + mt * 16 + vr + ((lane >> 4) << 3);
        int c = colBase + warpN * 32 + nt * 16 + (lane & 15);
        out[(size_t)r * 1024 + c] = acc[mt][nt][vr] + bc[c] + ws[c] * val[r];
      }
}

// ---------------- recurrent LSTM: one workgroup per (16-batch-half, direction) ----------------
// 512 threads = 16 waves; per step each wave: 4 N-tiles x 8 K-steps of wmma.

__global__ __launch_bounds__(512)
void k_lstm(const unsigned short* __restrict__ WhhBf,
            const float* __restrict__ preG,
            float* __restrict__ hbuf) {
  __shared__ unsigned short hS[16][264];           // h (bf16), padded
  __shared__ unsigned short gS[16][1032];          // hh-gemm result (bf16), padded
  __shared__ float cS[16][256];                    // cell state (f32)
  int tid = threadIdx.x, lane = tid & 31, w = tid >> 5;
  int bBase = blockIdx.x * 16;
  int dir = blockIdx.y;
  const unsigned short* Whh = WhhBf + (size_t)dir * (1024 * 256);
  const float* pre = preG + (size_t)dir * (8192ull * 1024ull);
  float* hout = hbuf + (size_t)dir * (8192ull * 256ull);

  for (int i = tid; i < 16 * 264; i += 512) ((unsigned short*)hS)[i] = 0;
  for (int i = tid; i < 16 * 256; i += 512) ((float*)cS)[i] = 0.f;
  __syncthreads();

  v8f zero = {0, 0, 0, 0, 0, 0, 0, 0};
  for (int t = 0; t < 256; ++t) {
    int s = dir ? (255 - t) : t;
    #pragma unroll
    for (int nt = 0; nt < 4; ++nt) {
      int col0 = w * 64 + nt * 16;                 // gate column tile base
      v8f acc = zero;
      #pragma unroll
      for (int ks = 0; ks < 8; ++ks) {
        FragBF a, b;
        const U4* pa = (const U4*)&hS[lane & 15][ks * 32];
        a.q[0] = pa[lane >> 4];
        a.q[1] = pa[2 + (lane >> 4)];
        const U4* pb = (const U4*)&Whh[(size_t)(col0 + (lane & 15)) * 256 + ks * 32];
        b.q[0] = pb[lane >> 4];
        b.q[1] = pb[2 + (lane >> 4)];
        acc = __builtin_amdgcn_wmma_f32_16x16x32_bf16(
            false, a.v, false, b.v, (short)0, acc, false, false);
      }
      #pragma unroll
      for (int vr = 0; vr < 8; ++vr) {
        int m = vr + ((lane >> 4) << 3);
        gS[m][col0 + (lane & 15)] = f2bf(acc[vr]);
      }
    }
    __syncthreads();
    // prefetch next step's pre-gate rows while doing the cell update
    if (t + 1 < 256) {
      int sn = dir ? (254 - t) : (t + 1);
      int m = tid & 15;
      __builtin_prefetch(pre + ((size_t)(bBase + m) * 256 + sn) * 1024 + (tid >> 4) * 32, 0, 1);
    }
    for (int i = tid; i < 16 * 256; i += 512) {    // cell update, 8 per thread
      int m = i >> 8, u = i & 255;
      size_t r = (size_t)(bBase + m) * 256 + s;
      const float* pg = pre + r * 1024;
      float ig = bf2f(gS[m][u])       + pg[u];
      float fg = bf2f(gS[m][256 + u]) + pg[256 + u];
      float gg = bf2f(gS[m][512 + u]) + pg[512 + u];
      float og = bf2f(gS[m][768 + u]) + pg[768 + u];
      float c = sigf(fg) * cS[m][u] + sigf(ig) * tanhf(gg);
      float h = sigf(og) * tanhf(c);
      cS[m][u] = c;
      hS[m][u] = f2bf(h);
      hout[r * 256 + u] = h;
    }
    __syncthreads();
  }
}

// ---------------- emissions + CE loss ----------------

__global__ void k_loss(const float* __restrict__ hf, const float* __restrict__ hb,
                       const float* __restrict__ Wt, const float* __restrict__ bt,
                       const int* __restrict__ labels, float* __restrict__ acc2) {
  int lane = threadIdx.x & 31, w = threadIdx.x >> 5;
  int r = blockIdx.x * 8 + w;                      // token id
  float a[9];
  #pragma unroll
  for (int t = 0; t < 9; ++t) a[t] = 0.f;
  for (int k = lane; k < 512; k += 32) {
    float hv = (k < 256) ? hf[(size_t)r * 256 + k] : hb[(size_t)r * 256 + (k - 256)];
    #pragma unroll
    for (int t = 0; t < 9; ++t) a[t] += hv * Wt[t * 512 + k];
  }
  #pragma unroll
  for (int t = 0; t < 9; ++t) a[t] = wred(a[t]);
  if (lane == 0) {
    float mx = -1e30f;
    #pragma unroll
    for (int t = 0; t < 9; ++t) { a[t] += bt[t]; mx = fmaxf(mx, a[t]); }
    float se = 0.f;
    #pragma unroll
    for (int t = 0; t < 9; ++t) se += __expf(a[t] - mx);
    float lse = mx + __logf(se);
    int lab = labels[r];
    if (lab != -100) {
      int l = lab < 0 ? 0 : (lab > 8 ? 8 : lab);
      atomicAdd(&acc2[0], lse - a[l]);
      atomicAdd(&acc2[1], 1.0f);
    }
  }
}

__global__ void k_final(const float* __restrict__ acc2, float* __restrict__ out) {
  if (threadIdx.x == 0) out[0] = acc2[0] / fmaxf(acc2[1], 1.0f);
}

// ---------------- launch ----------------

extern "C" void kernel_launch(void* const* d_in, const int* in_sizes, int n_in,
                              void* d_out, int out_size, void* d_ws, size_t ws_size,
                              hipStream_t stream) {
  const float* word   = (const float*)d_in[0];
  const float* chare  = (const float*)d_in[1];
  const int*   spans  = (const int*)d_in[2];
  const int*   labels = (const int*)d_in[3];
  const float* Wih_f  = (const float*)d_in[4];
  const float* Whh_f  = (const float*)d_in[5];
  const float* bih_f  = (const float*)d_in[6];
  const float* bhh_f  = (const float*)d_in[7];
  const float* Wih_b  = (const float*)d_in[8];
  const float* Whh_b  = (const float*)d_in[9];
  const float* bih_b  = (const float*)d_in[10];
  const float* bhh_b  = (const float*)d_in[11];
  const float* Wtag   = (const float*)d_in[12];
  const float* btag   = (const float*)d_in[13];
  float* out = (float*)d_out;
  (void)in_sizes; (void)n_in; (void)out_size; (void)ws_size;

  char* base = (char*)d_ws;
  size_t off = 0;
  auto alloc = [&](size_t bytes) -> void* {
    void* p = base + off;
    off = (off + bytes + 255) & ~(size_t)255;
    return p;
  };
  float*          featmean = (float*)alloc(32768ull * 4);
  float*          val      = (float*)alloc(8192ull * 4);
  float*          wsum     = (float*)alloc(2048ull * 4);
  float*          biasc    = (float*)alloc(2048ull * 4);
  float*          acc2     = (float*)alloc(2ull * 4);
  unsigned short* Xbf      = (unsigned short*)alloc(6291456ull * 2);
  unsigned short* WihBf    = (unsigned short*)alloc(2ull * 786432 * 2);
  unsigned short* WhhBf    = (unsigned short*)alloc(2ull * 262144 * 2);
  float*          preG     = (float*)alloc(2ull * 8388608 * 4);
  float*          hbuf     = (float*)alloc(2ull * 2097152 * 4);

  hipLaunchKernelGGL(k_zero,     dim3(1),          dim3(32),  0, stream, acc2);
  hipLaunchKernelGGL(k_featmean, dim3(4096),       dim3(256), 0, stream, chare, featmean);
  hipLaunchKernelGGL(k_val,      dim3(32),         dim3(256), 0, stream, spans, featmean, val);
  hipLaunchKernelGGL(k_cvt_x,    dim3(24576),      dim3(256), 0, stream, word, Xbf, 6291456);
  hipLaunchKernelGGL(k_prep_wih, dim3(256),        dim3(256), 0, stream,
                     Wih_f, Wih_b, bih_f, bhh_f, bih_b, bhh_b, WihBf, wsum, biasc);
  hipLaunchKernelGGL(k_cvt_whh,  dim3(2048),       dim3(256), 0, stream, Whh_f, Whh_b, WhhBf);
  hipLaunchKernelGGL(k_pregemm,  dim3(16, 64, 2),  dim3(256), 0, stream,
                     Xbf, WihBf, wsum, biasc, val, preG);
  hipLaunchKernelGGL(k_lstm,     dim3(2, 2),       dim3(512), 0, stream, WhhBf, preG, hbuf);
  hipLaunchKernelGGL(k_loss,     dim3(1024),       dim3(256), 0, stream,
                     hbuf, hbuf + 2097152ull, Wtag, btag, labels, acc2);
  hipLaunchKernelGGL(k_final,    dim3(1),          dim3(1),   0, stream, acc2, out);
}
